// ThalamicLSTM_24850680775007
// MI455X (gfx1250) — compile-verified, matrix operand order checked by reading, not run on hardware
//
#include <hip/hip_runtime.h>
#include <hip/hip_bf16.h>

// ---------------------------------------------------------------------------
// Persistent 2-layer LSTM for MI455X (gfx1250, wave32, WMMA).
//
//   - 4 batch groups of 16 rows (M = one 16x16 WMMA tile).
//   - 16 workgroups per group; WG w owns h-columns [32w, 32w+32)
//     => 8 gate tiles (2 per gate i/f/g/o), one per wave (8 waves / WG).
//   - Weight fragments for Wh0 / Wx1 / Wx0 are staged once into LDS
//     (264 KB of the 320 KB WGP LDS); only Wh1 streams from L2 per step.
//   - Recurrence h exchanged through a parity-double-buffered bf16 global
//     buffer; group-wide barrier (atomics in d_ws) twice per timestep.
//   - Cell state c kept in f32 in LDS (persistent across the t-loop).
// ---------------------------------------------------------------------------

typedef __bf16 bf16_t;
typedef bf16_t v16bf __attribute__((ext_vector_type(16)));
typedef bf16_t v8bf  __attribute__((ext_vector_type(8)));
typedef float  v8f   __attribute__((ext_vector_type(8)));

#define N_B      64
#define N_T      1024
#define N_IN     21
#define N_H      512
#define N_O      20
#define GROUPS   4
#define WPG      16            // workgroups per batch group
#define NWG      (GROUPS*WPG)  // 64
#define THREADS  256
#define H_STRIDE 512           // bf16 elems per row of h buffer
#define H_BUFELE (16*H_STRIDE) // one group's h slab (16 rows)

// ---- d_ws layout (bytes) ----
#define OFF_BAR   0u                                   // 4 groups * 32 uints
#define OFF_XF    512u                                 // 1024*4*512 bf16 = 4 MB
#define SZ_XF     (N_T*GROUPS*512u*2u)
#define OFF_WX0F  (OFF_XF + SZ_XF)                     // 16*8*1*512 bf16
#define SZ_WX0F   (WPG*8u*512u*2u)
#define OFF_WH0F  (OFF_WX0F + SZ_WX0F)                 // 16*8*16*512 bf16 = 2 MB
#define SZ_WHF    (WPG*8u*16u*512u*2u)
#define OFF_WX1F  (OFF_WH0F + SZ_WHF)
#define OFF_WH1F  (OFF_WX1F + SZ_WHF)
#define OFF_WOF   (OFF_WH1F + SZ_WHF)                  // 2*16*512 bf16
#define SZ_WOF    (2u*16u*512u*2u)
#define OFF_H0    (OFF_WOF + SZ_WOF)                   // 2 parities * 4 groups * 16*512 bf16
#define SZ_H      (2u*GROUPS*H_BUFELE*2u)
#define OFF_H1    (OFF_H0 + SZ_H)

// ---- dynamic LDS layout (bytes) ----
#define LDSO_WH0  0u                       // 8*16*512 bf16 = 131072
#define LDSO_WX1  (LDSO_WH0 + 131072u)     // 131072
#define LDSO_WX0  (LDSO_WX1 + 131072u)     // 8*512 bf16 = 8192
#define LDSO_G    (LDSO_WX0 + 8192u)       // 8*256 f32 = 8192
#define LDSO_C0   (LDSO_G + 8192u)         // 512 f32 = 2048
#define LDSO_C1   (LDSO_C0 + 2048u)        // 2048
#define LDS_BYTES (LDSO_C1 + 2048u)        // 282624 B (< 320 KB WGP LDS)

// ---------------------------------------------------------------------------
__device__ inline v8f wmma_bf16(v16bf a, v16bf b, v8f c) {
  return __builtin_amdgcn_wmma_f32_16x16x32_bf16(false, a, false, b, (short)0,
                                                 c, false, false);
}

// A-fragment (16x32 bf16) loaded straight from a row-major bf16 h buffer
// (row stride H_STRIDE). ISA 16-bit A layout: lanes 0-15: M=lane,
// K = {kb..kb+7, kb+16..kb+23}; lanes 16-31: +8 K offset.
__device__ inline v16bf load_afrag(const bf16_t* __restrict__ base,
                                   unsigned lane, unsigned kbase) {
  const unsigned row  = lane & 15u;
  const unsigned koff = (lane >> 4) * 8u;
  const bf16_t* p = base + (size_t)row * H_STRIDE + kbase + koff;
  v8bf lo = *(const v8bf*)(p);
  v8bf hi = *(const v8bf*)(p + 16);
  v16bf a;
#pragma unroll
  for (int i = 0; i < 8; ++i) { a[i] = lo[i]; a[i + 8] = hi[i]; }
  return a;
}

// Prepacked fragment: one contiguous 32B load per lane (global or LDS).
__device__ inline v16bf load_pfrag(const bf16_t* buf, unsigned fragIdx,
                                   unsigned lane) {
  return *(const v16bf*)(buf + ((size_t)fragIdx * 32u + lane) * 16u);
}

__device__ inline float fast_sigmoid(float x) {
  return __builtin_amdgcn_rcpf(1.0f + __expf(-x));
}
__device__ inline float fast_tanh(float x) {
  x = fminf(fmaxf(x, -20.0f), 20.0f);
  float e = __expf(-2.0f * x);
  return (1.0f - e) * __builtin_amdgcn_rcpf(1.0f + e);
}

// Generation-counter barrier across the WPG workgroups of one batch group.
__device__ inline void group_barrier(unsigned* cnt, unsigned* gen) {
  __syncthreads();
  if (threadIdx.x == 0) {
    __threadfence();
    unsigned g = __hip_atomic_load(gen, __ATOMIC_RELAXED, __HIP_MEMORY_SCOPE_AGENT);
    unsigned a = __hip_atomic_fetch_add(cnt, 1u, __ATOMIC_ACQ_REL, __HIP_MEMORY_SCOPE_AGENT);
    if (a == WPG - 1u) {
      __hip_atomic_store(cnt, 0u, __ATOMIC_RELAXED, __HIP_MEMORY_SCOPE_AGENT);
      __hip_atomic_fetch_add(gen, 1u, __ATOMIC_RELEASE, __HIP_MEMORY_SCOPE_AGENT);
    } else {
      while (__hip_atomic_load(gen, __ATOMIC_ACQUIRE, __HIP_MEMORY_SCOPE_AGENT) == g)
        __builtin_amdgcn_s_sleep(1);
    }
    __threadfence();
  }
  __syncthreads();
}

// ---------------------------------------------------------------------------
// Prep kernels (re-run every launch; deterministic).
// ---------------------------------------------------------------------------

// K index pattern shared by A and B fragment packing (16-bit WMMA layout).
__device__ inline unsigned kpat(unsigned lane, unsigned e) {
  return (lane >> 4) * 8u + (e < 8u ? e : e + 8u);
}

__global__ void k_init(bf16_t* h0, bf16_t* h1, unsigned* bar) {
  unsigned idx = blockIdx.x * blockDim.x + threadIdx.x;   // 65536 threads
  h0[idx] = (bf16_t)0.0f;
  h1[idx] = (bf16_t)0.0f;
  if (idx < GROUPS * 32u) bar[idx] = 0u;
}

// Pack a (2048 x 512) weight into B-fragments: frag (w, j, kk):
// n = (j>>1)*512 + w*32 + (j&1)*16 + (lane&15); k = kk*32 + kpat.
__global__ void k_pack_wh(const float* __restrict__ W, bf16_t* __restrict__ dst) {
  unsigned idx  = blockIdx.x * blockDim.x + threadIdx.x;  // 1,048,576
  unsigned e    = idx & 15u;
  unsigned lane = (idx >> 4) & 31u;
  unsigned kk   = (idx >> 9) & 15u;
  unsigned j    = (idx >> 13) & 7u;
  unsigned w    = idx >> 16;
  unsigned n = (j >> 1) * 512u + w * 32u + (j & 1u) * 16u + (lane & 15u);
  unsigned k = kk * 32u + kpat(lane, e);
  dst[idx] = (bf16_t)W[(size_t)n * 512u + k];
}

// Wx0 (2048 x 21), K padded to 32: frag (w, j).
__global__ void k_pack_wx0(const float* __restrict__ W, bf16_t* __restrict__ dst) {
  unsigned idx  = blockIdx.x * blockDim.x + threadIdx.x;  // 65,536
  unsigned e    = idx & 15u;
  unsigned lane = (idx >> 4) & 31u;
  unsigned j    = (idx >> 9) & 7u;
  unsigned w    = idx >> 12;
  unsigned n = (j >> 1) * 512u + w * 32u + (j & 1u) * 16u + (lane & 15u);
  unsigned k = kpat(lane, e);
  dst[idx] = (k < N_IN) ? (bf16_t)W[(size_t)n * N_IN + k] : (bf16_t)0.0f;
}

// Wo (20 x 512), N padded to 32: frag (j, kk), j in {0,1}.
__global__ void k_pack_wo(const float* __restrict__ W, bf16_t* __restrict__ dst) {
  unsigned idx  = blockIdx.x * blockDim.x + threadIdx.x;  // 16,384
  unsigned e    = idx & 15u;
  unsigned lane = (idx >> 4) & 31u;
  unsigned kk   = (idx >> 9) & 15u;
  unsigned j    = idx >> 13;
  unsigned n = j * 16u + (lane & 15u);
  unsigned k = kk * 32u + kpat(lane, e);
  dst[idx] = (n < N_O) ? (bf16_t)W[(size_t)n * N_H + k] : (bf16_t)0.0f;
}

// x (64,1024,21) -> A-fragments per (t, group): K padded to 32.
__global__ void k_pack_x(const float* __restrict__ x, bf16_t* __restrict__ dst) {
  unsigned idx  = blockIdx.x * blockDim.x + threadIdx.x;  // 2,097,152
  unsigned e    = idx & 15u;
  unsigned lane = (idx >> 4) & 31u;
  unsigned g    = (idx >> 9) & 3u;
  unsigned t    = idx >> 11;
  unsigned row  = g * 16u + (lane & 15u);
  unsigned k    = kpat(lane, e);
  dst[idx] = (k < N_IN) ? (bf16_t)x[((size_t)row * N_T + t) * N_IN + k]
                        : (bf16_t)0.0f;
}

// ---------------------------------------------------------------------------
// Persistent LSTM kernel: 64 WGs x 256 threads, 276 KB LDS each.
// ---------------------------------------------------------------------------
__global__ __launch_bounds__(THREADS, 1)
void k_lstm(const float* __restrict__ bx0, const float* __restrict__ bx1,
            const float* __restrict__ bo,
            const bf16_t* __restrict__ XF,
            const bf16_t* __restrict__ WX0F, const bf16_t* __restrict__ WH0F,
            const bf16_t* __restrict__ WX1F, const bf16_t* __restrict__ WH1F,
            const bf16_t* __restrict__ WOF,
            bf16_t* __restrict__ H0, bf16_t* __restrict__ H1,
            unsigned* __restrict__ bar,
            float* __restrict__ out) {
  const unsigned tid  = threadIdx.x;
  const unsigned g    = blockIdx.x / WPG;   // batch group 0..3
  const unsigned w    = blockIdx.x % WPG;   // WG within group
  const unsigned wave = tid >> 5;
  const unsigned lane = tid & 31u;
  const unsigned j    = wave;               // gate tile 0..7 (gate=j>>1, sub=j&1)
  const unsigned hb   = w * 32u;            // owned h-column base
  const unsigned gcol = (j >> 1) * 512u + hb + (j & 1u) * 16u + (lane & 15u);
  const unsigned mbase = (lane >> 4) * 8u;  // C-tile row base for this lane

  extern __shared__ char smem[];
  bf16_t* ldsWH0 = (bf16_t*)(smem + LDSO_WH0);  // this WG's Wh0 frags (128 KB)
  bf16_t* ldsWX1 = (bf16_t*)(smem + LDSO_WX1);  // this WG's Wx1 frags (128 KB)
  bf16_t* ldsWX0 = (bf16_t*)(smem + LDSO_WX0);  // this WG's Wx0 frags (8 KB)
  float*  ldsG   = (float*)(smem + LDSO_G);     // 8 gate tiles (16x16 f32)
  float*  ldsC0  = (float*)(smem + LDSO_C0);    // cell state, layer 0 slice
  float*  ldsC1  = (float*)(smem + LDSO_C1);    // cell state, layer 1 slice

  // ---- one-time staging: pin layer-0 weights + Wx1 in LDS ----
  {
    const uint4* s0 = (const uint4*)(WH0F + (size_t)w * 65536u);
    const uint4* s1 = (const uint4*)(WX1F + (size_t)w * 65536u);
    uint4* d0 = (uint4*)ldsWH0;
    uint4* d1 = (uint4*)ldsWX1;
    for (unsigned i = tid; i < 8192u; i += THREADS) { d0[i] = s0[i]; d1[i] = s1[i]; }
    const uint4* s2 = (const uint4*)(WX0F + (size_t)w * 4096u);
    uint4* d2 = (uint4*)ldsWX0;
    for (unsigned i = tid; i < 256u; i += THREADS) d2[i] = s2[i];
  }
  for (unsigned i = tid; i < 512u; i += THREADS) { ldsC0[i] = 0.0f; ldsC1[i] = 0.0f; }
  __syncthreads();

  unsigned* cnt = bar + g * 32u;
  unsigned* gen = bar + g * 32u + 16u;

  const size_t fb = ((size_t)w * 8u + j) * 16u;  // global frag base (Wh1)

  for (unsigned t = 0; t < N_T; ++t) {
    const unsigned pw = t & 1u, pr = pw ^ 1u;
    const bf16_t* h0r = H0 + (size_t)(pr * GROUPS + g) * H_BUFELE;
    bf16_t*       h0w = H0 + (size_t)(pw * GROUPS + g) * H_BUFELE;
    const bf16_t* h1r = H1 + (size_t)(pr * GROUPS + g) * H_BUFELE;
    bf16_t*       h1w = H1 + (size_t)(pw * GROUPS + g) * H_BUFELE;

    // ---------------- layer 0: gates = x@Wx0^T + h0@Wh0^T + bx0 ----------
    {
      v8f acc = {};
      {
        v16bf ax = load_pfrag(XF, t * GROUPS + g, lane);
        v16bf bx = load_pfrag(ldsWX0, j, lane);
        acc = wmma_bf16(ax, bx, acc);
      }
#pragma unroll 4
      for (unsigned kk = 0; kk < 16u; ++kk) {
        v16bf ah = load_afrag(h0r, lane, kk * 32u);
        v16bf bw = load_pfrag(ldsWH0, j * 16u + kk, lane);
        acc = wmma_bf16(ah, bw, acc);
      }
      const float bias = bx0[gcol];
#pragma unroll
      for (int r = 0; r < 8; ++r)
        ldsG[j * 256u + (mbase + r) * 16u + (lane & 15u)] = acc[r] + bias;
    }
    __syncthreads();
    // h0/c0 elementwise update: 2 of the 16x32 values per thread.
    {
      const unsigned row = tid >> 4;
      const unsigned c2  = (tid & 15u) * 2u;
#pragma unroll
      for (unsigned q = 0; q < 2u; ++q) {
        unsigned cc = c2 + q;
        unsigned s = cc >> 4, nn = cc & 15u;
        float iG = ldsG[(0u + s) * 256u + row * 16u + nn];
        float fG = ldsG[(2u + s) * 256u + row * 16u + nn];
        float gG = ldsG[(4u + s) * 256u + row * 16u + nn];
        float oG = ldsG[(6u + s) * 256u + row * 16u + nn];
        float c_new = fast_sigmoid(fG) * ldsC0[row * 32u + cc] +
                      fast_sigmoid(iG) * fast_tanh(gG);
        ldsC0[row * 32u + cc] = c_new;
        h0w[(size_t)row * H_STRIDE + hb + cc] =
            (bf16_t)(fast_sigmoid(oG) * fast_tanh(c_new));
      }
    }
    group_barrier(cnt, gen);

    // ---------------- layer 1: gates = h0@Wx1^T + h1@Wh1^T + bx1 ---------
    {
      v8f acc = {};
#pragma unroll 4
      for (unsigned kk = 0; kk < 16u; ++kk) {
        v16bf ah = load_afrag(h0w, lane, kk * 32u);
        v16bf bw = load_pfrag(ldsWX1, j * 16u + kk, lane);
        acc = wmma_bf16(ah, bw, acc);
      }
#pragma unroll 4
      for (unsigned kk = 0; kk < 16u; ++kk) {
        v16bf ah = load_afrag(h1r, lane, kk * 32u);
        v16bf bw = load_pfrag(WH1F, (unsigned)(fb + kk), lane);
        acc = wmma_bf16(ah, bw, acc);
      }
      const float bias = bx1[gcol];
#pragma unroll
      for (int r = 0; r < 8; ++r)
        ldsG[j * 256u + (mbase + r) * 16u + (lane & 15u)] = acc[r] + bias;
    }
    __syncthreads();
    {
      const unsigned row = tid >> 4;
      const unsigned c2  = (tid & 15u) * 2u;
#pragma unroll
      for (unsigned q = 0; q < 2u; ++q) {
        unsigned cc = c2 + q;
        unsigned s = cc >> 4, nn = cc & 15u;
        float iG = ldsG[(0u + s) * 256u + row * 16u + nn];
        float fG = ldsG[(2u + s) * 256u + row * 16u + nn];
        float gG = ldsG[(4u + s) * 256u + row * 16u + nn];
        float oG = ldsG[(6u + s) * 256u + row * 16u + nn];
        float c_new = fast_sigmoid(fG) * ldsC1[row * 32u + cc] +
                      fast_sigmoid(iG) * fast_tanh(gG);
        ldsC1[row * 32u + cc] = c_new;
        h1w[(size_t)row * H_STRIDE + hb + cc] =
            (bf16_t)(fast_sigmoid(oG) * fast_tanh(c_new));
      }
    }
    group_barrier(cnt, gen);

    // ---------------- output head: out = sigmoid(h1@Wo^T + bo) ----------
    if (w == 0u && j < 2u) {
      v8f acc = {};
#pragma unroll 4
      for (unsigned kk = 0; kk < 16u; ++kk) {
        v16bf ah = load_afrag(h1w, lane, kk * 32u);
        v16bf bw = load_pfrag(WOF, j * 16u + kk, lane);
        acc = wmma_bf16(ah, bw, acc);
      }
      const unsigned n = j * 16u + (lane & 15u);
      if (n < N_O) {
        const float bias = bo[n];
#pragma unroll
        for (int r = 0; r < 8; ++r) {
          unsigned grow = g * 16u + mbase + r;
          out[((size_t)grow * N_T + t) * N_O + n] = fast_sigmoid(acc[r] + bias);
        }
      }
    }
  }
}

// ---------------------------------------------------------------------------
extern "C" void kernel_launch(void* const* d_in, const int* in_sizes, int n_in,
                              void* d_out, int out_size, void* d_ws, size_t ws_size,
                              hipStream_t stream) {
  (void)in_sizes; (void)n_in; (void)out_size; (void)ws_size;
  const float* x   = (const float*)d_in[0];
  const float* Wx0 = (const float*)d_in[1];
  const float* bx0 = (const float*)d_in[2];
  const float* Wh0 = (const float*)d_in[3];
  const float* Wx1 = (const float*)d_in[4];
  const float* bx1 = (const float*)d_in[5];
  const float* Wh1 = (const float*)d_in[6];
  const float* Wo  = (const float*)d_in[7];
  const float* bo  = (const float*)d_in[8];
  float* out = (float*)d_out;

  char* ws = (char*)d_ws;
  unsigned* bar   = (unsigned*)(ws + OFF_BAR);
  bf16_t*   XF    = (bf16_t*)(ws + OFF_XF);
  bf16_t*   WX0F  = (bf16_t*)(ws + OFF_WX0F);
  bf16_t*   WH0F  = (bf16_t*)(ws + OFF_WH0F);
  bf16_t*   WX1F  = (bf16_t*)(ws + OFF_WX1F);
  bf16_t*   WH1F  = (bf16_t*)(ws + OFF_WH1F);
  bf16_t*   WOF   = (bf16_t*)(ws + OFF_WOF);
  bf16_t*   H0    = (bf16_t*)(ws + OFF_H0);
  bf16_t*   H1    = (bf16_t*)(ws + OFF_H1);

  k_init<<<256, THREADS, 0, stream>>>(H0, H1, bar);
  k_pack_wh<<<4096, THREADS, 0, stream>>>(Wh0, WH0F);
  k_pack_wh<<<4096, THREADS, 0, stream>>>(Wx1, WX1F);
  k_pack_wh<<<4096, THREADS, 0, stream>>>(Wh1, WH1F);
  k_pack_wx0<<<256, THREADS, 0, stream>>>(Wx0, WX0F);
  k_pack_wo<<<64, THREADS, 0, stream>>>(Wo, WOF);
  k_pack_x<<<8192, THREADS, 0, stream>>>(x, XF);

  k_lstm<<<NWG, THREADS, LDS_BYTES, stream>>>(bx0, bx1, bo, XF, WX0F, WH0F,
                                              WX1F, WH1F, WOF, H0, H1, bar, out);
}
